// AttentionalCopula_24438363915036
// MI455X (gfx1250) — matrix operand order repaired and checked by reading, not compiled, not run on hardware
//
#include <hip/hip_runtime.h>
#include <hip/hip_bf16.h>

// ---------------------------------------------------------------------------
// Problem constants (match reference)
// ---------------------------------------------------------------------------
#define BB  4
#define VN  1024          // NS*NT
#define HN  2048
#define WW  3072          // HN+VN
#define DD  256
#define LL  2
#define NH  4
#define AA  32
#define HD  128           // NH*AA
#define MLP 128
#define RES 128
#define KPAD 288          // 257 padded up to multiple of 32
#define EPSLN 1e-5f

typedef unsigned short u16;
typedef __bf16 bf16_t;
typedef bf16_t v16bf __attribute__((ext_vector_type(16)));
typedef u16    v16u  __attribute__((ext_vector_type(16)));
typedef float  v8f   __attribute__((ext_vector_type(8)));
typedef int    v4i   __attribute__((ext_vector_type(4)));
typedef int    v8i   __attribute__((ext_vector_type(8)));

// ---------------------------------------------------------------------------
// Helpers
// ---------------------------------------------------------------------------
__device__ __forceinline__ u16 f2bf(float x) {
  unsigned int u = __float_as_uint(x);
  unsigned int r = u + 0x7FFFu + ((u >> 16) & 1u);   // round-to-nearest-even
  return (u16)(r >> 16);
}

__device__ __forceinline__ v8f wmma_bf16(v16bf a, v16bf b, v8f c) {
  return __builtin_amdgcn_wmma_f32_16x16x32_bf16(false, a, false, b,
                                                 (short)0, c, false, false);
}

// A-fragment: 16x32 bf16 (MxK), row-major source, row stride ld (elements).
__device__ __forceinline__ v16bf load_a_frag(const u16* p, int ld) {
  int lane = threadIdx.x & 31;
  const u16* row = p + (lane & 15) * ld;
  int kb  = (lane < 16) ? 0 : 8;
  int kb2 = kb + 16;
  v16u r;
#pragma unroll
  for (int i = 0; i < 8; ++i) r[i] = row[kb + i];
#pragma unroll
  for (int i = 0; i < 8; ++i) r[8 + i] = row[kb2 + i];
  return __builtin_bit_cast(v16bf, r);
}

// B-fragment: 32x16 bf16 (KxN), row-major source, row stride ld. lane%16 = N column.
__device__ __forceinline__ v16bf load_b_frag(const u16* p, int ld) {
  int lane = threadIdx.x & 31;
  const u16* col = p + (lane & 15);
  int kb  = (lane < 16) ? 0 : 8;
  int kb2 = kb + 16;
  v16u r;
#pragma unroll
  for (int i = 0; i < 8; ++i) r[i] = col[(kb + i) * ld];
#pragma unroll
  for (int i = 0; i < 8; ++i) r[8 + i] = col[(kb2 + i) * ld];
  return __builtin_bit_cast(v16bf, r);
}

// ---------------------------------------------------------------------------
// Tensor Data Mover: 2-D tile (bf16 elements) from global -> LDS.
// Builds the D# descriptor per CDNA5 ISA:
//   group0: count=1 | lds_addr | global_addr | type=2
//   group1: data_size=1(2B), tensor_dim0/1, tile_dim0/1, tensor_dim0_stride
//   group2/3: zero (dims 2..4 unused; tile_dim2=0)
// Issued via inline asm (portable across toolchains); tracked by TENSORcnt.
// ---------------------------------------------------------------------------
__device__ __forceinline__ void tdm_load_2d(unsigned lds_addr, const u16* gptr,
                                            unsigned dim0, unsigned dim1,
                                            unsigned tile0, unsigned tile1,
                                            unsigned long long stride0) {
  unsigned long long ga = (unsigned long long)(uintptr_t)gptr;
  v4i g0; v8i g1; v4i g2 = {0, 0, 0, 0}; v4i g3 = {0, 0, 0, 0};
  g0[0] = 1;                                   // count=1, user mode
  g0[1] = (int)lds_addr;                       // lds_addr (bytes)
  g0[2] = (int)(unsigned)ga;                   // global_addr[31:0]
  g0[3] = (int)(((unsigned)(ga >> 32) & 0x01FFFFFFu) | 0x80000000u); // [56:32] | type=2
  g1[0] = (1 << 16);                           // data_size = 1 -> 2 bytes
  g1[1] = (int)((dim0 & 0xFFFFu) << 16);       // tensor_dim0[15:0]  @ bits 63:48
  g1[2] = (int)((dim0 >> 16) | ((dim1 & 0xFFFFu) << 16)); // dim0[31:16], dim1[15:0]
  g1[3] = (int)((dim1 >> 16) | (tile0 << 16)); // dim1[31:16], tile_dim0
  g1[4] = (int)(tile1 & 0xFFFFu);              // tile_dim1 ; tile_dim2 = 0
  g1[5] = (int)(unsigned)stride0;              // tensor_dim0_stride[31:0]
  g1[6] = (int)((unsigned)(stride0 >> 32) & 0xFFFFu); // stride[47:32]; dim1_stride lo = 0
  g1[7] = 0;
  asm volatile("tensor_load_to_lds %0, %1, %2, %3"
               :: "s"(g0), "s"(g1), "s"(g2), "s"(g3)
               : "memory");
}

// ---------------------------------------------------------------------------
// Conversion / packing kernels
// ---------------------------------------------------------------------------
__global__ void cvt_f32_bf16(const float* __restrict__ s, u16* __restrict__ d, int n) {
  int i = blockIdx.x * blockDim.x + threadIdx.x;
  if (i < n) d[i] = f2bf(s[i]);
}

__global__ void build_kvin(const float* __restrict__ he, const float* __restrict__ hu,
                           const float* __restrict__ pe, const float* __restrict__ pu,
                           u16* __restrict__ d) {
  int i = blockIdx.x * blockDim.x + threadIdx.x;
  const int total = BB * WW * KPAD;
  if (i >= total) return;
  int k = i % KPAD; int t = i / KPAD;
  int w = t % WW;   int b = t / WW;
  float v = 0.f;
  if (k < DD) {
    v = (w < HN) ? he[(b * HN + w) * DD + k] : pe[(b * VN + (w - HN)) * DD + k];
  } else if (k == DD) {
    v = (w < HN) ? hu[b * HN + w] : pu[b * VN + (w - HN)];
  }
  d[i] = f2bf(v);
}

__global__ void build_kvw(const float* __restrict__ kc, const float* __restrict__ vc,
                          u16* __restrict__ d) {
  int i = blockIdx.x * blockDim.x + threadIdx.x;
  const int total = LL * 2 * NH * KPAD * AA;
  if (i >= total) return;
  int a = i % AA; int t = i / AA;
  int k = t % KPAD; t /= KPAD;
  int h = t % NH;  t /= NH;
  int kv = t & 1;  int l = t >> 1;
  float v = 0.f;
  if (k < DD + 1) {
    const float* s = kv ? vc : kc;
    v = s[((l * NH + h) * (DD + 1) + k) * AA + a];
  }
  d[i] = f2bf(v);
}

// ---------------------------------------------------------------------------
// Generic WMMA GEMM: one wave per 16x16 tile, K multiple of 32.
// ---------------------------------------------------------------------------
__global__ void __launch_bounds__(256)
gemm_bf16(const u16* __restrict__ A, const u16* __restrict__ Bm,
          const float* __restrict__ bias, float* __restrict__ C,
          u16* __restrict__ Cbf, int M, int N, int K, int relu) {
  int wave = (int)((blockIdx.x * blockDim.x + threadIdx.x) >> 5);
  int tilesN = N >> 4;
  int tm = (wave / tilesN) << 4;
  int tn = (wave % tilesN) << 4;
  if (tm >= M) return;
  v8f acc = {};
  for (int k0 = 0; k0 < K; k0 += 32) {
    v16bf af = load_a_frag(A + tm * K + k0, K);
    v16bf bf = load_b_frag(Bm + k0 * N + tn, N);
    acc = wmma_bf16(af, bf, acc);
  }
  int lane = threadIdx.x & 31;
  int n = tn + (lane & 15);
  float bs = bias ? bias[n] : 0.f;
  int moff = tm + ((lane < 16) ? 0 : 8);
#pragma unroll
  for (int r = 0; r < 8; ++r) {
    float v = acc[r] + bs;
    if (relu) v = fmaxf(v, 0.f);
    int m = moff + r;
    C[m * N + n] = v;
    if (Cbf) Cbf[m * N + n] = f2bf(v);
  }
}

// ---------------------------------------------------------------------------
// K/V projection -> keysT [L][B][NH][A][W] and vals [L][B][NH][W][A]
// ---------------------------------------------------------------------------
__global__ void __launch_bounds__(256)
kvproj(const u16* __restrict__ kvin, const u16* __restrict__ wkv,
       const float* __restrict__ kc_b, const float* __restrict__ vc_b,
       u16* __restrict__ keysT, u16* __restrict__ vals) {
  int wave = (int)((blockIdx.x * blockDim.x + threadIdx.x) >> 5);
  int tn = wave & 1;  wave >>= 1;
  int tm = wave % (WW / 16); wave /= (WW / 16);
  int h  = wave % NH; wave /= NH;
  int b  = wave % BB; wave /= BB;
  int kv = wave & 1;  int l = wave >> 1;
  if (l >= LL) return;

  const u16* Ap = kvin + (b * WW + tm * 16) * KPAD;
  const u16* Wp = wkv + (size_t)(((l * 2 + kv) * NH + h)) * KPAD * AA + tn * 16;
  v8f acc = {};
  for (int k0 = 0; k0 < KPAD; k0 += 32) {
    v16bf af = load_a_frag(Ap + k0, KPAD);
    v16bf bf = load_b_frag(Wp + k0 * AA, AA);
    acc = wmma_bf16(af, bf, acc);
  }
  int lane = threadIdx.x & 31;
  int n = tn * 16 + (lane & 15);
  float bias = (kv ? vc_b : kc_b)[(l * NH + h) * AA + n];
  int moff = tm * 16 + ((lane < 16) ? 0 : 8);
#pragma unroll
  for (int r = 0; r < 8; ++r) {
    int m = moff + r;
    u16 bv = f2bf(acc[r] + bias);
    if (kv == 0)
      keysT[((size_t)((l * BB + b) * NH + h) * AA + n) * WW + m] = bv;
    else
      vals [((size_t)((l * BB + b) * NH + h) * WW + m) * AA + n] = bv;
  }
}

// ---------------------------------------------------------------------------
// Flash attention, 4 waves/block, one wave per (b, h, 16-query tile).
// K/V 32x32 bf16 chunks are DMA'd into LDS by the Tensor Data Mover with
// double buffering; the wave consumes the previous chunk with WMMA while the
// next chunk is in flight (TENSORcnt-tracked).
// ---------------------------------------------------------------------------
__global__ void __launch_bounds__(128)
flash_attn(const u16* __restrict__ attv_bf, const u16* __restrict__ keysT,
           const u16* __restrict__ vals, float* __restrict__ attout, int l) {
  __shared__ float Ssh[4][16 * 32];
  __shared__ u16   Psh[4][16 * 32];
  __shared__ float Stat[4][48];
  __shared__ u16   Ktile[4][2][32 * 32];
  __shared__ u16   Vtile[4][2][32 * 32];

  int widx = threadIdx.x >> 5;
  int lane = threadIdx.x & 31;
  int wg = (int)(blockIdx.x * 4 + widx);
  int qt = wg & 63;  wg >>= 6;
  int h  = wg & 3;   int b = wg >> 2;
  if (b >= BB) return;
  int v0 = qt * 16;

  const u16* kT = keysT + (size_t)((l * BB + b) * NH + h) * AA * WW; // [A][W]
  const u16* vp = vals  + (size_t)((l * BB + b) * NH + h) * WW * AA; // [W][A]
  v16bf qf = load_a_frag(attv_bf + (size_t)(b * VN + v0) * HD + h * AA, HD);

  float* S = Ssh[widx];
  u16*   P = Psh[widx];
  float* rowmax = Stat[widx];
  float* rowsum = rowmax + 16;
  float* corr   = rowmax + 32;
  if (lane < 16) { rowmax[lane] = -1e30f; rowsum[lane] = 0.f; }
  asm volatile("" ::: "memory");

  v8f o0 = {}, o1 = {};
  const float scale = 0.17677669529663687f; // 32^-0.5
  int wlimit = HN + v0 + 15;
  int nch = (wlimit + 31) >> 5;

  // prefetch chunk 0 via TDM
  tdm_load_2d((unsigned)(uintptr_t)&Ktile[widx][0][0], kT, WW, AA, 32, 32, WW);
  tdm_load_2d((unsigned)(uintptr_t)&Vtile[widx][0][0], vp, AA, WW, 32, 32, AA);

  for (int ci = 0; ci < nch; ++ci) {
    int w0 = ci << 5;
    int cur = ci & 1;
    if (ci + 1 < nch) {
      // buffer (ci+1)&1 was consumed in iteration ci-1: drain its DS reads
      asm volatile("s_wait_dscnt 0x0" ::: "memory");
      int nxt = (ci + 1) & 1;
      tdm_load_2d((unsigned)(uintptr_t)&Ktile[widx][nxt][0], kT + (w0 + 32),
                  WW, AA, 32, 32, WW);
      tdm_load_2d((unsigned)(uintptr_t)&Vtile[widx][nxt][0], vp + (size_t)(w0 + 32) * AA,
                  AA, WW, 32, 32, AA);
      __builtin_amdgcn_s_wait_tensorcnt(2);   // chunk ci complete, next in flight
    } else {
      __builtin_amdgcn_s_wait_tensorcnt(0);
    }
    asm volatile("" ::: "memory");

    const u16* ldsK = &Ktile[widx][cur][0];   // [a][w_local], ld=32
    const u16* ldsV = &Vtile[widx][cur][0];   // [w_local][a], ld=32

    v16bf kb0 = load_b_frag(ldsK, 32);
    v16bf kb1 = load_b_frag(ldsK + 16, 32);
    v8f z = {};
    v8f s0 = wmma_bf16(qf, kb0, z);
    v8f s1 = wmma_bf16(qf, kb1, z);

    int n = lane & 15;
    int moff = (lane < 16) ? 0 : 8;
#pragma unroll
    for (int r = 0; r < 8; ++r) {
      int m = r + moff;
      int vglob = v0 + m;
      float sv0 = s0[r] * scale;
      float sv1 = s1[r] * scale;
      if (!((w0 + n)      < HN + vglob)) sv0 = -1e30f;
      if (!((w0 + 16 + n) < HN + vglob)) sv1 = -1e30f;
      S[m * 32 + n]      = sv0;
      S[m * 32 + 16 + n] = sv1;
    }
    asm volatile("" ::: "memory");

    if (lane < 16) {
      int m = lane;
      float oldm = rowmax[m];
      float cm = -1e30f;
#pragma unroll
      for (int j = 0; j < 32; ++j) cm = fmaxf(cm, S[m * 32 + j]);
      float newm = fmaxf(oldm, cm);
      float c = __expf(oldm - newm);
      float sum = rowsum[m] * c;
#pragma unroll
      for (int j = 0; j < 32; ++j) {
        float p = __expf(S[m * 32 + j] - newm);
        sum += p;
        P[m * 32 + j] = f2bf(p);
      }
      rowmax[m] = newm; rowsum[m] = sum; corr[m] = c;
    }
    asm volatile("" ::: "memory");

#pragma unroll
    for (int r = 0; r < 8; ++r) {
      float c = corr[r + moff];
      o0[r] *= c;
      o1[r] *= c;
    }
    v16bf pf  = load_a_frag(P, 32);
    v16bf vb0 = load_b_frag(ldsV, 32);
    v16bf vb1 = load_b_frag(ldsV + 16, 32);
    o0 = wmma_bf16(pf, vb0, o0);
    o1 = wmma_bf16(pf, vb1, o1);
  }

  asm volatile("" ::: "memory");
  int n = lane & 15;
  int moff = (lane < 16) ? 0 : 8;
#pragma unroll
  for (int r = 0; r < 8; ++r) {
    int m = r + moff;
    float inv = 1.f / rowsum[m];
    size_t base = (size_t)(b * VN + v0 + m) * HD + h * AA;
    attout[base + n]      = o0[r] * inv;
    attout[base + 16 + n] = o1[r] * inv;
  }
}

// ---------------------------------------------------------------------------
// out = LayerNorm(x + res) * g + b ; also bf16 mirror. One wave per row.
// ---------------------------------------------------------------------------
__global__ void __launch_bounds__(256)
add_ln(const float* __restrict__ x, const float* __restrict__ res,
       const float* __restrict__ g, const float* __restrict__ bta,
       float* __restrict__ out, u16* __restrict__ outbf) {
  int row = (int)((blockIdx.x * blockDim.x + threadIdx.x) >> 5);
  if (row >= BB * VN) return;
  int lane = threadIdx.x & 31;
  const float* xr = x + (size_t)row * HD;
  const float* rr = res ? res + (size_t)row * HD : nullptr;
  float v[4], s = 0.f;
#pragma unroll
  for (int i = 0; i < 4; ++i) {
    float t = xr[lane + 32 * i] + (rr ? rr[lane + 32 * i] : 0.f);
    v[i] = t; s += t;
  }
#pragma unroll
  for (int o = 16; o; o >>= 1) s += __shfl_xor(s, o, 32);
  float mean = s * (1.f / HD);
  float var = 0.f;
#pragma unroll
  for (int i = 0; i < 4; ++i) { float d = v[i] - mean; var += d * d; }
#pragma unroll
  for (int o = 16; o; o >>= 1) var += __shfl_xor(var, o, 32);
  float inv = rsqrtf(var * (1.f / HD) + EPSLN);
#pragma unroll
  for (int i = 0; i < 4; ++i) {
    int c = lane + 32 * i;
    float y = (v[i] - mean) * inv * g[c] + bta[c];
    out[(size_t)row * HD + c] = y;
    outbf[(size_t)row * HD + c] = f2bf(y);
  }
}

// ---------------------------------------------------------------------------
// NLL reduction
// ---------------------------------------------------------------------------
__global__ void __launch_bounds__(256)
nll_reduce(const float* __restrict__ logits, const float* __restrict__ pu,
           float* __restrict__ out) {
  __shared__ float red[256];
  int b = blockIdx.x;
  float acc = 0.f;
  for (int v = 1 + (int)threadIdx.x; v < VN; v += (int)blockDim.x) {
    const float* row = logits + (size_t)(b * VN + v) * RES;
    int t = (int)floorf(pu[b * VN + v] * (float)RES);
    t = min(max(t, 0), RES - 1);
    float mx = -1e30f;
    for (int j = 0; j < RES; ++j) mx = fmaxf(mx, row[j]);
    float s = 0.f;
    for (int j = 0; j < RES; ++j) s += __expf(row[j] - mx);
    acc += 4.852030263919617f /*log 128*/ + row[t] - mx - __logf(s);
  }
  red[threadIdx.x] = acc;
  __syncthreads();
  for (int o = 128; o; o >>= 1) {
    if ((int)threadIdx.x < o) red[threadIdx.x] += red[threadIdx.x + o];
    __syncthreads();
  }
  if (threadIdx.x == 0) out[b] = -red[0];
}

// ---------------------------------------------------------------------------
// Host orchestration
// ---------------------------------------------------------------------------
static inline int blocks_for(long long threads) { return (int)((threads + 255) / 256); }

extern "C" void kernel_launch(void* const* d_in, const int* in_sizes, int n_in,
                              void* d_out, int out_size, void* d_ws, size_t ws_size,
                              hipStream_t stream) {
  (void)in_sizes; (void)n_in; (void)out_size; (void)ws_size;
  const float* hist_e  = (const float*)d_in[0];
  const float* hist_u  = (const float*)d_in[1];
  const float* pred_e  = (const float*)d_in[2];
  const float* pred_u  = (const float*)d_in[3];
  const float* W_shift = (const float*)d_in[4];
  const float* b_shift = (const float*)d_in[5];
  const float* ln_att_g = (const float*)d_in[6];
  const float* ln_att_b = (const float*)d_in[7];
  const float* ln_ff_g  = (const float*)d_in[8];
  const float* ln_ff_b  = (const float*)d_in[9];
  const float* kc_W = (const float*)d_in[10];
  const float* kc_b = (const float*)d_in[11];
  const float* vc_W = (const float*)d_in[12];
  const float* vc_b = (const float*)d_in[13];
  const float* ff1_W = (const float*)d_in[14];
  const float* ff1_b = (const float*)d_in[15];
  const float* ff2_W = (const float*)d_in[16];
  const float* ff2_b = (const float*)d_in[17];
  const float* ff3_W = (const float*)d_in[18];
  const float* ff3_b = (const float*)d_in[19];
  const float* dist_W = (const float*)d_in[20];
  const float* dist_b = (const float*)d_in[21];
  float* out = (float*)d_out;

  char* ws = (char*)d_ws;
  size_t off = 0;
  auto take = [&](size_t bytes) { char* p = ws + off; off += (bytes + 255) & ~(size_t)255; return p; };
  u16*   kvin_bf  = (u16*)take((size_t)BB * WW * KPAD * 2);
  u16*   kvw_bf   = (u16*)take((size_t)LL * 2 * NH * KPAD * AA * 2);
  u16*   predA_bf = (u16*)take((size_t)BB * VN * DD * 2);
  u16*   wsh_bf   = (u16*)take((size_t)DD * HD * 2);
  u16*   ffw_bf   = (u16*)take((size_t)3 * LL * MLP * MLP * 2);
  u16*   dw_bf    = (u16*)take((size_t)HD * RES * 2);
  u16*   keysT    = (u16*)take((size_t)LL * BB * NH * AA * WW * 2);
  u16*   vals     = (u16*)take((size_t)LL * BB * NH * WW * AA * 2);
  float* attv     = (float*)take((size_t)BB * VN * HD * 4);
  u16*   attv_bf  = (u16*)take((size_t)BB * VN * HD * 2);
  float* attout   = (float*)take((size_t)BB * VN * HD * 4);
  float* fbuf1    = (float*)take((size_t)BB * VN * MLP * 4);
  u16*   fbuf1_bf = (u16*)take((size_t)BB * VN * MLP * 2);
  float* fbuf2    = (float*)take((size_t)BB * VN * MLP * 4);
  u16*   fbuf2_bf = (u16*)take((size_t)BB * VN * MLP * 2);
  float* logits   = (float*)take((size_t)BB * VN * RES * 4);

  {
    int n = BB * VN * DD;
    cvt_f32_bf16<<<blocks_for(n), 256, 0, stream>>>(pred_e, predA_bf, n);
  }
  cvt_f32_bf16<<<blocks_for(DD * HD), 256, 0, stream>>>(W_shift, wsh_bf, DD * HD);
  cvt_f32_bf16<<<blocks_for(LL * MLP * MLP), 256, 0, stream>>>(ff1_W, ffw_bf + 0 * LL * MLP * MLP, LL * MLP * MLP);
  cvt_f32_bf16<<<blocks_for(LL * MLP * MLP), 256, 0, stream>>>(ff2_W, ffw_bf + 1 * LL * MLP * MLP, LL * MLP * MLP);
  cvt_f32_bf16<<<blocks_for(LL * MLP * MLP), 256, 0, stream>>>(ff3_W, ffw_bf + 2 * LL * MLP * MLP, LL * MLP * MLP);
  cvt_f32_bf16<<<blocks_for(HD * RES), 256, 0, stream>>>(dist_W, dw_bf, HD * RES);
  build_kvin<<<blocks_for((long long)BB * WW * KPAD), 256, 0, stream>>>(hist_e, hist_u, pred_e, pred_u, kvin_bf);
  build_kvw<<<blocks_for((long long)LL * 2 * NH * KPAD * AA), 256, 0, stream>>>(kc_W, vc_W, kvw_bf);

  {
    int tiles = (BB * VN / 16) * (HD / 16);
    gemm_bf16<<<blocks_for((long long)tiles * 32), 256, 0, stream>>>(
        predA_bf, wsh_bf, b_shift, attv, attv_bf, BB * VN, HD, DD, 0);
  }
  {
    long long waves = (long long)LL * 2 * BB * NH * (WW / 16) * 2;
    kvproj<<<blocks_for(waves * 32), 256, 0, stream>>>(kvin_bf, kvw_bf, kc_b, vc_b, keysT, vals);
  }

  for (int l = 0; l < LL; ++l) {
    {
      int waves = BB * NH * (VN / 16);           // 1024 waves, 4 per block
      flash_attn<<<waves / 4, 128, 0, stream>>>(attv_bf, keysT, vals, attout, l);
    }
    add_ln<<<blocks_for((long long)BB * VN * 32), 256, 0, stream>>>(
        attout, attv, ln_att_g + l * HD, ln_att_b + l * HD, attv, attv_bf);

    int tiles = (BB * VN / 16) * (MLP / 16);
    gemm_bf16<<<blocks_for((long long)tiles * 32), 256, 0, stream>>>(
        attv_bf, ffw_bf + (0 * LL + l) * MLP * MLP, ff1_b + l * MLP,
        fbuf1, fbuf1_bf, BB * VN, MLP, HD, 1);
    gemm_bf16<<<blocks_for((long long)tiles * 32), 256, 0, stream>>>(
        fbuf1_bf, ffw_bf + (1 * LL + l) * MLP * MLP, ff2_b + l * MLP,
        fbuf2, fbuf2_bf, BB * VN, MLP, MLP, 1);
    gemm_bf16<<<blocks_for((long long)tiles * 32), 256, 0, stream>>>(
        fbuf2_bf, ffw_bf + (2 * LL + l) * MLP * MLP, ff3_b + l * MLP,
        fbuf1, (u16*)nullptr, BB * VN, HD, MLP, 0);
    add_ln<<<blocks_for((long long)BB * VN * 32), 256, 0, stream>>>(
        fbuf1, attv, ln_ff_g + l * HD, ln_ff_b + l * HD, attv, attv_bf);
  }

  {
    int tiles = (BB * VN / 16) * (RES / 16);
    gemm_bf16<<<blocks_for((long long)tiles * 32), 256, 0, stream>>>(
        attv_bf, dw_bf, dist_b, logits, (u16*)nullptr, BB * VN, RES, HD, 0);
  }
  nll_reduce<<<BB, 256, 0, stream>>>(logits, pred_u, out);
}
